// MDLSTM_38989713113281
// MI455X (gfx1250) — matrix operand order repaired and checked by reading, not compiled, not run on hardware
//
#include <hip/hip_runtime.h>

#define HH 32
#define WW 128
#define BB 32
#define CINN 64
#define OCC 128
#define NGG 640   // 5*OC

typedef __attribute__((ext_vector_type(16))) __bf16 v16bf;
typedef __attribute__((ext_vector_type(8)))  float  v8f;
typedef __attribute__((ext_vector_type(8)))  unsigned short u16x8;
typedef __attribute__((ext_vector_type(16))) unsigned short u16x16;

// -------- workspace byte offsets (all 256B aligned), total ~24.7 MB --------
#define OFF_XB   0u          // x in bf16, layout (pix=y*W+x, b, c): 4096*32*64*2 = 16,777,216 B
#define OFF_W0T  16777216u   // w0^T bf16, per dir (g=640, c=64):   4*640*64*2  = 327,680 B
#define OFF_U0T  17104896u   // u0^T bf16, per dir (g=640, o=128):  4*640*128*2 = 655,360 B
#define OFF_U1T  17760256u   // u1^T bf16
#define OFF_HD   18415616u   // h diagonal ping-pong bf16: 4 * 2*32*32*128 * 2 = 2,097,152 B
#define OFF_CD   20512768u   // c diagonal ping-pong f32 : 4 * 2*32*32*128 * 4 = 4,194,304 B

#define LDS_U_ELEMS (NGG * OCC)          // 81,920 ushorts per u-array
#define LDS_BYTES   (2 * LDS_U_ELEMS * 2) // 327,680 B = 320 KB

__device__ __forceinline__ unsigned short f2bf(float f) {
  unsigned int u = __builtin_bit_cast(unsigned int, f);
  u += 0x7FFFu + ((u >> 16) & 1u);            // round-to-nearest-even
  return (unsigned short)(u >> 16);
}

__device__ __forceinline__ float fsig(float x)  { return 1.f / (1.f + __expf(-x)); }
__device__ __forceinline__ float ftanh(float x) { return 1.f - 2.f / (__expf(2.f * x) + 1.f); }

// A fragment (16x32 bf16, M x K): lane holds row M=lane&15; elements 0..15 = K[off..off+15],
// where off = kbase + 16*(lane>=16).
__device__ __forceinline__ v16bf load_a_frag(const unsigned short* __restrict__ row, int off) {
  u16x8 lo = *(const u16x8*)(row + off);
  u16x8 hi = *(const u16x8*)(row + off + 8);
  u16x16 t = __builtin_shufflevector(lo, hi, 0,1,2,3,4,5,6,7,8,9,10,11,12,13,14,15);
  return __builtin_bit_cast(v16bf, t);
}

// B fragment from global (w0^T): lane holds column N=lane&15, 16 contiguous K values.
__device__ __forceinline__ v16bf load_b_frag_g(const unsigned short* __restrict__ p) {
  u16x16 t = *(const u16x16*)p;
  return __builtin_bit_cast(v16bf, t);
}

// B fragment from LDS with XOR-swizzled 32B chunks: element offset of (col, chunk) is
// col*128 + (chunk ^ (col&7))*16.  col&7 == (lane&15)&7 for our columns.
__device__ __forceinline__ v16bf load_b_frag_lds(const unsigned short* p, int col, int chunk) {
  const unsigned short* q = p + col * OCC + ((chunk ^ (col & 7)) * 16);
  u16x16 t = *(const u16x16*)q;
  return __builtin_bit_cast(v16bf, t);
}

// -------------------- prep: weights -> bf16, transposed --------------------
__global__ void prep_weights(const float* __restrict__ w0, const float* __restrict__ u0,
                             const float* __restrict__ u1, unsigned char* __restrict__ wsb) {
  unsigned short* w0T = (unsigned short*)(wsb + OFF_W0T);
  unsigned short* u0T = (unsigned short*)(wsb + OFF_U0T);
  unsigned short* u1T = (unsigned short*)(wsb + OFF_U1T);
  const int n_w0 = 4 * NGG * CINN;           // 163,840
  const int n_u  = 4 * NGG * OCC;            // 327,680
  const int total = n_w0 + 2 * n_u;          // 819,200
  for (int t = blockIdx.x * blockDim.x + threadIdx.x; t < total; t += gridDim.x * blockDim.x) {
    if (t < n_w0) {
      int d = t / (NGG * CINN), r = t % (NGG * CINN), g = r / CINN, c = r % CINN;
      w0T[t] = f2bf(w0[(d * CINN + c) * NGG + g]);
    } else if (t < n_w0 + n_u) {
      int t2 = t - n_w0;
      int d = t2 / (NGG * OCC), r = t2 % (NGG * OCC), g = r / OCC, o = r % OCC;
      u0T[t2] = f2bf(u0[(d * OCC + o) * NGG + g]);
    } else {
      int t2 = t - n_w0 - n_u;
      int d = t2 / (NGG * OCC), r = t2 % (NGG * OCC), g = r / OCC, o = r % OCC;
      u1T[t2] = f2bf(u1[(d * OCC + o) * NGG + g]);
    }
  }
}

// -------------------- prep: x -> bf16, pixel-major (pix,b,c) --------------------
__global__ void prep_x(const float* __restrict__ x, unsigned char* __restrict__ wsb) {
  unsigned short* xb = (unsigned short*)(wsb + OFF_XB);
  const int total = BB * CINN * HH * WW;     // 8,388,608
  for (int t = blockIdx.x * blockDim.x + threadIdx.x; t < total; t += gridDim.x * blockDim.x) {
    int c   = t & (CINN - 1);
    int b   = (t >> 6) & 31;
    int pix = t >> 11;
    int y   = pix >> 7;          // W = 128
    int xw  = pix & 127;
    xb[t] = f2bf(x[((b * CINN + c) * HH + y) * WW + xw]);
  }
}

// -------------------- main recurrence: 1 workgroup per direction --------------------
// 512 threads = 16 waves; wave w owns wavefront slots {w, w+16}.
// Unit = (slot, o-tile): both 16-row batch halves computed together sharing B fragments.
__global__ __launch_bounds__(512, 1)
void mdlstm_scan(unsigned char* __restrict__ wsb, const float* __restrict__ bias_all,
                 float* __restrict__ out) {
  extern __shared__ unsigned short smem[];   // [2][640][128] swizzled u0^T, u1^T

  const int dir  = blockIdx.x;
  const int tid  = threadIdx.x;
  const int wave = tid >> 5;
  const int lane = tid & 31;
  const int n16  = lane & 15;
  const int half = lane >> 4;
  const int aoff = half * 16;      // A-fragment per-lane K base
  const int boff = half * 16;      // B-fragment per-lane K base (global w0T path)
  const int Mb   = half * 8;       // C-fragment per-lane M base

  const unsigned short* xb  = (const unsigned short*)(wsb + OFF_XB);
  const unsigned short* w0T = (const unsigned short*)(wsb + OFF_W0T) + dir * (NGG * CINN);
  const unsigned short* u0T = (const unsigned short*)(wsb + OFF_U0T) + dir * (NGG * OCC);
  const unsigned short* u1T = (const unsigned short*)(wsb + OFF_U1T) + dir * (NGG * OCC);
  unsigned short* hd = (unsigned short*)(wsb + OFF_HD) + dir * (2 * HH * BB * OCC);
  float*          cd = (float*)(wsb + OFF_CD) + dir * (2 * HH * BB * OCC);
  const float* bias = bias_all + dir * NGG;

  // ---- stage u0^T, u1^T into LDS (XOR-swizzled 32B chunks) ----
  for (int idx = tid; idx < 2 * NGG * 8; idx += blockDim.x) {
    const int arr   = idx / (NGG * 8);
    const int r     = idx % (NGG * 8);
    const int col   = r >> 3;
    const int chunk = r & 7;
    const unsigned short* src = (arr ? u1T : u0T) + col * OCC + chunk * 16;
    unsigned short* dst = smem + arr * LDS_U_ELEMS + col * OCC + ((chunk ^ (col & 7)) * 16);
    *(u16x8*)dst       = *(const u16x8*)src;
    *(u16x8*)(dst + 8) = *(const u16x8*)(src + 8);
  }
  __syncthreads();

  for (int step = 0; step < HH + WW - 1; ++step) {
    const int cur  = step & 1;
    const int prev = cur ^ 1;
    const int y0 = (step > WW - 1) ? (step - (WW - 1)) : 0;

    for (int sl = 0; sl < 2; ++sl) {
      const int slot = wave + sl * 16;
      const int y = y0 + slot;
      const int x = step - y;
      const bool valid = (y < HH) && (x >= 0);
      if (!valid) continue;

      const int xx = (dir & 1) ? (WW - 1 - x) : x;   // flip width
      const int yy = (dir & 2) ? (HH - 1 - y) : y;   // flip height
      const bool m_up = (y > 0);
      const bool m_lf = (x > 0);
      const int yu = m_up ? (y - 1) : 0;
      const unsigned short* hup_base = hd + ((prev * HH + yu) * BB) * OCC;
      const unsigned short* hlf_base = hd + ((prev * HH + y) * BB) * OCC;
      const float* cup_base = cd + ((prev * HH + yu) * BB) * OCC;
      const float* clf_base = cd + ((prev * HH + y) * BB) * OCC;
      unsigned short* hcur_base = hd + ((cur * HH + y) * BB) * OCC;
      float*          ccur_base = cd + ((cur * HH + y) * BB) * OCC;
      const unsigned short* xpix = xb + ((yy * WW + xx) * BB) * CINN;

      const unsigned short* ax_row0  = xpix     + n16 * CINN;
      const unsigned short* ax_row1  = xpix     + (16 + n16) * CINN;
      const unsigned short* aup_row0 = hup_base + n16 * OCC;
      const unsigned short* aup_row1 = hup_base + (16 + n16) * OCC;
      const unsigned short* alf_row0 = hlf_base + n16 * OCC;
      const unsigned short* alf_row1 = hlf_base + (16 + n16) * OCC;

      for (int ot = 0; ot < 8; ++ot) {     // 16-wide o-tile
        v8f acc[2][5];
        #pragma unroll
        for (int g = 0; g < 5; ++g) {
          const float bv = bias[g * OCC + ot * 16 + n16];
          v8f t = {bv, bv, bv, bv, bv, bv, bv, bv};
          acc[0][g] = t;
          acc[1][g] = t;
        }
        // x @ w0 (K = 64), B from global (hot in WGP$)
        #pragma unroll
        for (int kk = 0; kk < CINN; kk += 32) {
          const v16bf a0 = load_a_frag(ax_row0, kk + aoff);
          const v16bf a1 = load_a_frag(ax_row1, kk + aoff);
          #pragma unroll
          for (int g = 0; g < 5; ++g) {
            const v16bf bf = load_b_frag_g(w0T + (g * OCC + ot * 16 + n16) * CINN + kk + boff);
            acc[0][g] = __builtin_amdgcn_wmma_f32_16x16x32_bf16(false, a0, false, bf,
                                                                (short)0, acc[0][g], false, false);
            acc[1][g] = __builtin_amdgcn_wmma_f32_16x16x32_bf16(false, a1, false, bf,
                                                                (short)0, acc[1][g], false, false);
          }
        }
        // h_up @ u0 (K = 128), B from LDS
        if (m_up) {
          #pragma unroll
          for (int kk = 0; kk < OCC; kk += 32) {
            const v16bf a0 = load_a_frag(aup_row0, kk + aoff);
            const v16bf a1 = load_a_frag(aup_row1, kk + aoff);
            #pragma unroll
            for (int g = 0; g < 5; ++g) {
              const v16bf bf = load_b_frag_lds(smem, g * OCC + ot * 16 + n16, (kk >> 4) + half);
              acc[0][g] = __builtin_amdgcn_wmma_f32_16x16x32_bf16(false, a0, false, bf,
                                                                  (short)0, acc[0][g], false, false);
              acc[1][g] = __builtin_amdgcn_wmma_f32_16x16x32_bf16(false, a1, false, bf,
                                                                  (short)0, acc[1][g], false, false);
            }
          }
        }
        // h_lf @ u1 (K = 128), B from LDS
        if (m_lf) {
          #pragma unroll
          for (int kk = 0; kk < OCC; kk += 32) {
            const v16bf a0 = load_a_frag(alf_row0, kk + aoff);
            const v16bf a1 = load_a_frag(alf_row1, kk + aoff);
            #pragma unroll
            for (int g = 0; g < 5; ++g) {
              const v16bf bf = load_b_frag_lds(smem + LDS_U_ELEMS,
                                               g * OCC + ot * 16 + n16, (kk >> 4) + half);
              acc[0][g] = __builtin_amdgcn_wmma_f32_16x16x32_bf16(false, a0, false, bf,
                                                                  (short)0, acc[0][g], false, false);
              acc[1][g] = __builtin_amdgcn_wmma_f32_16x16x32_bf16(false, a1, false, bf,
                                                                  (short)0, acc[1][g], false, false);
            }
          }
        }
        // fused LSTM cell update for both batch halves, entirely in registers
        const int o = ot * 16 + n16;
        #pragma unroll
        for (int j = 0; j < 2; ++j) {
          #pragma unroll
          for (int v = 0; v < 8; ++v) {
            const int brow = j * 16 + Mb + v;
            const float cup = m_up ? cup_base[brow * OCC + o] : 0.f;
            const float clf = m_lf ? clf_base[brow * OCC + o] : 0.f;
            const float ig = fsig(acc[j][0][v]);
            const float fg = fsig(acc[j][1][v]);
            const float gg = ftanh(acc[j][2][v]);
            const float og = fsig(acc[j][3][v]);
            const float lg = fsig(acc[j][4][v]);
            const float cn = fg * (lg * cup + (1.f - lg) * clf) + ig * gg;
            const float hn = og * ftanh(cn);
            ccur_base[brow * OCC + o] = cn;
            hcur_base[brow * OCC + o] = f2bf(hn);
            // out layout: (B, 4, OC, H, W), flipped frame (no unflip, per reference)
            out[(((brow * 4 + dir) * OCC + o) * HH + y) * WW + x] = hn;
          }
        }
      }
    }
    __threadfence_block();   // make diag-buffer stores visible within the WGP
    __syncthreads();         // wavefront step barrier
  }
}

extern "C" void kernel_launch(void* const* d_in, const int* in_sizes, int n_in,
                              void* d_out, int out_size, void* d_ws, size_t ws_size,
                              hipStream_t stream) {
  const float* x  = (const float*)d_in[0];
  const float* w0 = (const float*)d_in[1];
  const float* u0 = (const float*)d_in[2];
  const float* u1 = (const float*)d_in[3];
  const float* b  = (const float*)d_in[4];
  unsigned char* wsb = (unsigned char*)d_ws;
  float* out = (float*)d_out;

  // allow 320 KB dynamic LDS (CDNA5 WGP limit); idempotent + capture-safe
  (void)hipFuncSetAttribute((const void*)mdlstm_scan,
                            hipFuncAttributeMaxDynamicSharedMemorySize, LDS_BYTES);

  prep_weights<<<1600, 256, 0, stream>>>(w0, u0, u1, wsb);
  prep_x<<<8192, 256, 0, stream>>>(x, wsb);
  mdlstm_scan<<<4, 512, LDS_BYTES, stream>>>(wsb, b, out);
}
